// SGJointModel_18537078850198
// MI455X (gfx1250) — matrix operand order repaired and checked by reading, not compiled, not run on hardware
//
#include <hip/hip_runtime.h>
#include <hip/hip_bf16.h>

typedef __attribute__((ext_vector_type(16))) _Float16 v16h;
typedef __attribute__((ext_vector_type(8)))  _Float16 v8h;
typedef __attribute__((ext_vector_type(8)))  float    v8f;

#define B_   32
#define S_   512
#define V_   30000
#define E_   512
#define H_   256
#define H2_  512
#define L_   2
#define G4_  1024           // 4*H
#define N2_  2048           // 2 dirs * 4H
#define SLOT_OUT_ 128
#define INTENT_OUT_ 64

// ---------------------------------------------------------------------------
// WMMA helpers (CDNA5 wave32, V_WMMA_F32_16X16X32_F16)
// ---------------------------------------------------------------------------
__device__ __forceinline__ v8f wmma_f16(v16h a, v16h b, v8f c) {
    // (neg_a, A, neg_b, B, c_mod, C, reuse_a, reuse_b)
    return __builtin_amdgcn_wmma_f32_16x16x32_f16(false, a, false, b,
                                                  (short)0, c, false, false);
}

// A-matrix 16x32 f16 fragment. lane L(0..15) holds row M=L,
// contiguous K[hi*16 .. hi*16+15] where hi=lane>>4 (matches load_b_frag).
__device__ __forceinline__ v16h load_a_frag(const _Float16* __restrict__ A,
                                            int lda, int m0, int k0) {
    int lane = threadIdx.x & 31;
    int hi   = lane >> 4;
    const _Float16* p = A + (size_t)(m0 + (lane & 15)) * lda + k0 + hi * 16;
    v8h lo  = *(const v8h*)(p);
    v8h hi8 = *(const v8h*)(p + 8);
    return __builtin_shufflevector(lo, hi8, 0,1,2,3,4,5,6,7,8,9,10,11,12,13,14,15);
}

// B-matrix 32x16 f16 fragment, weight stored (N,K) row-major so column n of B
// == row n of W: per-lane contiguous K chunk. lanes0-15: K=0..15, 16-31: K=16..31.
__device__ __forceinline__ v16h load_b_frag(const _Float16* __restrict__ W,
                                            int ldw, int n0, int k0) {
    int lane = threadIdx.x & 31;
    const _Float16* p = W + (size_t)(n0 + (lane & 15)) * ldw + k0 + ((lane >> 4) << 4);
    return *(const v16h*)p;
}

// ---------------------------------------------------------------------------
// f32 -> f16 conversion (weights)
// ---------------------------------------------------------------------------
__global__ void k_f2h(const float* __restrict__ src, _Float16* __restrict__ dst, long n) {
    long i = (long)blockIdx.x * blockDim.x + threadIdx.x;
    if (i < n) dst[i] = (_Float16)src[i];
}

// ---------------------------------------------------------------------------
// Embedding gather -> f16
// ---------------------------------------------------------------------------
__global__ void k_embed(const int* __restrict__ tok, const float* __restrict__ emb,
                        _Float16* __restrict__ x16) {
    long i = (long)blockIdx.x * blockDim.x + threadIdx.x;
    if (i >= (long)B_ * S_ * E_) return;
    long row = i / E_;
    int  col = (int)(i % E_);
    int  v   = tok[row];
    x16[i] = (_Float16)emb[(long)v * E_ + col];
}

// ---------------------------------------------------------------------------
// Generic (optionally batched) GEMM:  C = A(MxK,f16) * W^T (W is NxK f16) + bias
// block = 256 threads (8 waves): wave tile 32x32 (2x2 WMMA tiles),
// block tile 64x128, grid = (ceil(N/128), ceil(M/64), batch)
// ---------------------------------------------------------------------------
__global__ __launch_bounds__(256)
void k_gemm_f16(const _Float16* __restrict__ A, int lda, long strideA,
                const _Float16* __restrict__ W, int ldw, long strideW,
                float* __restrict__ C, int ldc, long strideC,
                const float* __restrict__ bias0, const float* __restrict__ bias1,
                int M, int N, int K)
{
    int bz = blockIdx.z;
    A += (long)bz * strideA;
    W += (long)bz * strideW;
    C += (long)bz * strideC;

    int wave = threadIdx.x >> 5;
    int lane = threadIdx.x & 31;
    int wm = wave >> 2, wn = wave & 3;
    int m0 = blockIdx.y * 64 + wm * 32;
    int n0 = blockIdx.x * 128 + wn * 32;
    if (m0 >= M || n0 >= N) return;

    v8f acc[2][2] = {};
    for (int k0 = 0; k0 < K; k0 += 32) {
        v16h a0 = load_a_frag(A, lda, m0,      k0);
        v16h a1 = load_a_frag(A, lda, m0 + 16, k0);
        v16h w0 = load_b_frag(W, ldw, n0,      k0);
        v16h w1 = load_b_frag(W, ldw, n0 + 16, k0);
        acc[0][0] = wmma_f16(a0, w0, acc[0][0]);
        acc[0][1] = wmma_f16(a0, w1, acc[0][1]);
        acc[1][0] = wmma_f16(a1, w0, acc[1][0]);
        acc[1][1] = wmma_f16(a1, w1, acc[1][1]);
    }

    int mo = (lane >> 4) * 8;   // C layout: m = tile + i + 8*(lane>=16)
    int nc = lane & 15;         // n = lane & 15
    for (int tm = 0; tm < 2; ++tm) {
        for (int tn = 0; tn < 2; ++tn) {
            int ng = n0 + tn * 16 + nc;
            float bvv = 0.f;
            if (bias0) bvv += bias0[ng];
            if (bias1) bvv += bias1[ng];
            int mg0 = m0 + tm * 16 + mo;
            for (int i = 0; i < 8; ++i) {
                C[(long)(mg0 + i) * ldc + ng] = acc[tm][tn][i] + bvv;
            }
        }
    }
}

// ---------------------------------------------------------------------------
// BiLSTM recurrence for one layer. grid = {2} (dir 0 fwd, dir 1 bwd),
// block = 512 threads (16 waves). Each wave owns a 16-wide h slice and
// computes all 4 gates for it -> wave-local cell update, c stays in VGPRs.
// pre layout: (B, S, 2, 4H) fp32, already includes bih+bhh.
// ---------------------------------------------------------------------------
__global__ __launch_bounds__(512)
void k_lstm_scan(const float* __restrict__ pre,
                 const _Float16* __restrict__ Whh16,   // (2, 4H, H) this layer
                 float* __restrict__ outf,             // (B,S,H2) or nullptr
                 _Float16* __restrict__ out16)         // (B,S,H2)
{
    __shared__ __align__(32) _Float16 hbuf[2][B_][H_];  // double-buffered h (32 KB)

    int dir  = blockIdx.x;
    int wave = threadIdx.x >> 5;
    int lane = threadIdx.x & 31;
    const _Float16* Wd = Whh16 + (size_t)dir * G4_ * H_;

    for (int idx = threadIdx.x; idx < B_ * H_; idx += blockDim.x)
        hbuf[0][idx / H_][idx % H_] = (_Float16)0.f;
    __syncthreads();

    int jb = wave * 16;               // h-column slice base for this wave
    int mo = (lane >> 4) * 8;
    int nc = lane & 15;
    v8f c0 = {}, c1 = {};             // persistent cell state (rows 0-15 / 16-31)

    for (int t = 0; t < S_; ++t) {
        int tidx = dir ? (S_ - 1 - t) : t;
        int buf  = t & 1;

        v8f g[2][4] = {};             // [m-tile][gate]
        for (int k0 = 0; k0 < H_; k0 += 32) {
            v16h a0 = load_a_frag(&hbuf[buf][0][0], H_, 0,  k0);
            v16h a1 = load_a_frag(&hbuf[buf][0][0], H_, 16, k0);
            #pragma unroll
            for (int gi = 0; gi < 4; ++gi) {
                v16h w = load_b_frag(Wd, H_, gi * H_ + jb, k0);
                g[0][gi] = wmma_f16(a0, w, g[0][gi]);
                g[1][gi] = wmma_f16(a1, w, g[1][gi]);
            }
        }

        #pragma unroll
        for (int tm = 0; tm < 2; ++tm) {
            v8f& cc = tm ? c1 : c0;
            #pragma unroll
            for (int i = 0; i < 8; ++i) {
                int b = tm * 16 + mo + i;
                int j = jb + nc;
                long prow = ((long)b * S_ + tidx) * N2_ + (long)dir * G4_;
                float gi_ = g[tm][0][i] + pre[prow + 0 * H_ + j];
                float gf_ = g[tm][1][i] + pre[prow + 1 * H_ + j];
                float gg_ = g[tm][2][i] + pre[prow + 2 * H_ + j];
                float go_ = g[tm][3][i] + pre[prow + 3 * H_ + j];
                float si = 1.f / (1.f + __expf(-gi_));
                float sf = 1.f / (1.f + __expf(-gf_));
                float so = 1.f / (1.f + __expf(-go_));
                float cv = sf * cc[i] + si * tanhf(gg_);
                cc[i] = cv;
                float hv = so * tanhf(cv);
                hbuf[buf ^ 1][b][j] = (_Float16)hv;
                long oidx = ((long)b * S_ + tidx) * H2_ + (long)dir * H_ + j;
                if (outf) outf[oidx] = hv;
                out16[oidx] = (_Float16)hv;
            }
        }
        __syncthreads();
    }
}

// ---------------------------------------------------------------------------
// Row softmax over 512 cols, f32 in -> f16 out. One wave per row.
// ---------------------------------------------------------------------------
__global__ __launch_bounds__(256)
void k_softmax_rows(const float* __restrict__ X, _Float16* __restrict__ Y16) {
    long row = (long)blockIdx.x * 8 + (threadIdx.x >> 5);
    int  lane = threadIdx.x & 31;
    const float* x = X + row * S_;
    float vals[16];
    float m = -3.4e38f;
    #pragma unroll
    for (int i = 0; i < 16; ++i) { vals[i] = x[lane + 32 * i]; m = fmaxf(m, vals[i]); }
    for (int o = 16; o >= 1; o >>= 1) m = fmaxf(m, __shfl_xor(m, o, 32));
    float s = 0.f;
    #pragma unroll
    for (int i = 0; i < 16; ++i) { vals[i] = __expf(vals[i] - m); s += vals[i]; }
    for (int o = 16; o >= 1; o >>= 1) s += __shfl_xor(s, o, 32);
    float inv = 1.f / s;
    _Float16* y = Y16 + row * S_;
    #pragma unroll
    for (int i = 0; i < 16; ++i) y[lane + 32 * i] = (_Float16)(vals[i] * inv);
}

// ---------------------------------------------------------------------------
// Transpose (B,S,H2) f16 -> (B,H2,S) f16
// ---------------------------------------------------------------------------
__global__ void k_transpose16(const _Float16* __restrict__ src, _Float16* __restrict__ dst) {
    long i = (long)blockIdx.x * blockDim.x + threadIdx.x;
    if (i >= (long)B_ * S_ * H2_) return;
    long b = i / ((long)S_ * H2_);
    long r = i - b * (long)S_ * H2_;
    int  s = (int)(r / H2_);
    int  d = (int)(r % H2_);
    dst[b * (long)H2_ * S_ + (long)d * S_ + s] = src[i];
}

// ---------------------------------------------------------------------------
// Intent attention: q = out[:, -1, :], iw = softmax(q . out_s),
// intent_ctx = sum_s iw * out_s.  One block per batch.
// ---------------------------------------------------------------------------
__global__ __launch_bounds__(256)
void k_intent_attn(const float* __restrict__ out, float* __restrict__ q,
                   float* __restrict__ ictx, float* __restrict__ qpi)
{
    __shared__ float logits[S_];
    __shared__ float red[256];
    int b = blockIdx.x;
    int tid = threadIdx.x, wave = tid >> 5, lane = tid & 31;
    const float* ob = out + (size_t)b * S_ * H2_;
    const float* qb = ob + (size_t)(S_ - 1) * H2_;

    for (int s = wave; s < S_; s += 8) {
        const float* r = ob + (size_t)s * H2_;
        float acc = 0.f;
        for (int d = lane; d < H2_; d += 32) acc += qb[d] * r[d];
        for (int o = 16; o >= 1; o >>= 1) acc += __shfl_xor(acc, o, 32);
        if (lane == 0) logits[s] = acc;
    }
    __syncthreads();

    float m = -3.4e38f;
    for (int s = tid; s < S_; s += 256) m = fmaxf(m, logits[s]);
    red[tid] = m; __syncthreads();
    for (int st = 128; st >= 1; st >>= 1) {
        if (tid < st) red[tid] = fmaxf(red[tid], red[tid + st]);
        __syncthreads();
    }
    m = red[0]; __syncthreads();

    float sum = 0.f;
    for (int s = tid; s < S_; s += 256) { float e = __expf(logits[s] - m); logits[s] = e; sum += e; }
    red[tid] = sum; __syncthreads();
    for (int st = 128; st >= 1; st >>= 1) {
        if (tid < st) red[tid] += red[tid + st];
        __syncthreads();
    }
    float inv = 1.f / red[0];
    __syncthreads();

    for (int d = tid; d < H2_; d += 256) {
        float acc = 0.f;
        for (int s = 0; s < S_; ++s) acc += logits[s] * ob[(size_t)s * H2_ + d];
        acc *= inv;
        float qv = qb[d];
        q[(size_t)b * H2_ + d]    = qv;
        ictx[(size_t)b * H2_ + d] = acc;
        qpi[(size_t)b * H2_ + d]  = acc + qv;
    }
}

// icl = intent_ctx @ Wg^T + bg   (tiny: 32x512, K=512)
__global__ void k_icl(const float* __restrict__ ictx, const float* __restrict__ Wg,
                      const float* __restrict__ bg, float* __restrict__ icl) {
    int i = blockIdx.x * blockDim.x + threadIdx.x;
    if (i >= B_ * H2_) return;
    int b = i / H2_, n = i % H2_;
    const float* w = Wg + (long)n * H2_;
    const float* x = ictx + (long)b * H2_;
    float acc = bg[n];
    for (int k = 0; k < H2_; ++k) acc += w[k] * x[k];
    icl[i] = acc;
}

// wvs[k] = sum_n Wv[n,k]  (k<H2);  wvs[H2] = sum(bv)
// (sum over output dim of a linear layer is linear -> collapses the Wv GEMM)
__global__ void k_wvsum(const float* __restrict__ Wv, const float* __restrict__ bv,
                        float* __restrict__ wvs) {
    int k = blockIdx.x * blockDim.x + threadIdx.x;
    if (k < H2_) {
        float a = 0.f;
        for (int n = 0; n < H2_; ++n) a += Wv[(long)n * H2_ + k];
        wvs[k] = a;
    } else if (k == H2_) {
        float a = 0.f;
        for (int n = 0; n < H2_; ++n) a += bv[n];
        wvs[H2_] = a;
    }
}

// gate = tanh(slot_ctx + icl) . wvs + wvs[H2];  slot_in = gate*slot_ctx + out (f16)
// One wave per (b,t) row.
__global__ __launch_bounds__(256)
void k_gate_slotin(const float* __restrict__ slot_ctx, const float* __restrict__ icl,
                   const float* __restrict__ outf, const float* __restrict__ wvs,
                   _Float16* __restrict__ slot_in16)
{
    size_t row = (size_t)blockIdx.x * 8 + (threadIdx.x >> 5);
    int lane = threadIdx.x & 31;
    int b = (int)(row / S_);
    const float* sc = slot_ctx + row * H2_;
    const float* ob = outf + row * H2_;
    const float* ic = icl + (size_t)b * H2_;
    float acc = 0.f;
    #pragma unroll
    for (int i = 0; i < 16; ++i) {
        int d = lane + 32 * i;
        acc += tanhf(sc[d] + ic[d]) * wvs[d];
    }
    for (int o = 16; o >= 1; o >>= 1) acc += __shfl_xor(acc, o, 32);
    float gate = acc + wvs[H2_];
    _Float16* si = slot_in16 + row * H2_;
    #pragma unroll
    for (int i = 0; i < 16; ++i) {
        int d = lane + 32 * i;
        si[d] = (_Float16)(gate * sc[d] + ob[d]);
    }
}

// intent_vec = (intent_ctx + q) @ Wint^T + bint   (32x64, K=512)
__global__ void k_intent_vec(const float* __restrict__ qpi, const float* __restrict__ Wint,
                             const float* __restrict__ bint, float* __restrict__ iv) {
    int i = blockIdx.x * blockDim.x + threadIdx.x;
    if (i >= B_ * INTENT_OUT_) return;
    int b = i / INTENT_OUT_, n = i % INTENT_OUT_;
    const float* w = Wint + (long)n * H2_;
    const float* x = qpi + (long)b * H2_;
    float acc = bint[n];
    for (int k = 0; k < H2_; ++k) acc += w[k] * x[k];
    iv[i] = acc;
}

// broadcast intent_vec over S into second output region
__global__ void k_bcast_intent(const float* __restrict__ iv, float* __restrict__ o2) {
    long i = (long)blockIdx.x * blockDim.x + threadIdx.x;
    if (i >= (long)B_ * S_ * INTENT_OUT_) return;
    long b = i / ((long)S_ * INTENT_OUT_);
    int  n = (int)(i % INTENT_OUT_);
    o2[i] = iv[b * INTENT_OUT_ + n];
}

// ---------------------------------------------------------------------------
// Host launcher
// ---------------------------------------------------------------------------
extern "C" void kernel_launch(void* const* d_in, const int* in_sizes, int n_in,
                              void* d_out, int out_size, void* d_ws, size_t ws_size,
                              hipStream_t stream)
{
    const int*   tok   = (const int*)  d_in[0];
    const float* emb   = (const float*)d_in[2];
    const float* Wih   = (const float*)d_in[3];
    const float* Whh   = (const float*)d_in[4];
    const float* bih   = (const float*)d_in[5];
    const float* bhh   = (const float*)d_in[6];
    const float* Wg    = (const float*)d_in[7];
    const float* bg    = (const float*)d_in[8];
    const float* Wv    = (const float*)d_in[9];
    const float* bv    = (const float*)d_in[10];
    const float* Wslot = (const float*)d_in[11];
    const float* bslot = (const float*)d_in[12];
    const float* Wint  = (const float*)d_in[13];
    const float* bint  = (const float*)d_in[14];

    float* slot_out = (float*)d_out;                               // (B,S,128)
    float* intent_o = slot_out + (long)B_ * S_ * SLOT_OUT_;        // (B,S,64)

    char* base = (char*)d_ws;
    size_t off = 0;
    auto alloc = [&](size_t bytes) -> void* {
        void* p = base + off;
        off += (bytes + 255) & ~(size_t)255;
        return p;
    };

    _Float16* x16     = (_Float16*)alloc((size_t)B_ * S_ * E_ * 2);
    _Float16* Wih16   = (_Float16*)alloc((size_t)L_ * 2 * G4_ * E_ * 2);
    _Float16* Whh16   = (_Float16*)alloc((size_t)L_ * 2 * G4_ * H_ * 2);
    _Float16* Wslot16 = (_Float16*)alloc((size_t)SLOT_OUT_ * H2_ * 2);
    float*    pre     = (float*)   alloc((size_t)B_ * S_ * N2_ * 4);
    _Float16* out0_16 = (_Float16*)alloc((size_t)B_ * S_ * H2_ * 2);
    _Float16* out1_16 = (_Float16*)alloc((size_t)B_ * S_ * H2_ * 2);
    float*    outf    = (float*)   alloc((size_t)B_ * S_ * H2_ * 4);
    _Float16* out_t16 = (_Float16*)alloc((size_t)B_ * H2_ * S_ * 2);
    float*    scores  = (float*)   alloc((size_t)B_ * S_ * S_ * 4);
    _Float16* w16     = (_Float16*)alloc((size_t)B_ * S_ * S_ * 2);
    float*    slotctx = (float*)   alloc((size_t)B_ * S_ * H2_ * 4);
    _Float16* slotin  = (_Float16*)alloc((size_t)B_ * S_ * H2_ * 2);
    float*    qbuf    = (float*)   alloc((size_t)B_ * H2_ * 4);
    float*    ictx    = (float*)   alloc((size_t)B_ * H2_ * 4);
    float*    qpi     = (float*)   alloc((size_t)B_ * H2_ * 4);
    float*    icl     = (float*)   alloc((size_t)B_ * H2_ * 4);
    float*    wvs     = (float*)   alloc((size_t)(H2_ + 1) * 4);
    float*    iv      = (float*)   alloc((size_t)B_ * INTENT_OUT_ * 4);

    // --- weight conversion ---
    {
        long n1 = (long)L_ * 2 * G4_ * E_;
        k_f2h<<<dim3((n1 + 255) / 256), 256, 0, stream>>>(Wih, Wih16, n1);
        long n2 = (long)L_ * 2 * G4_ * H_;
        k_f2h<<<dim3((n2 + 255) / 256), 256, 0, stream>>>(Whh, Whh16, n2);
        long n3 = (long)SLOT_OUT_ * H2_;
        k_f2h<<<dim3((n3 + 255) / 256), 256, 0, stream>>>(Wslot, Wslot16, n3);
    }

    // --- embedding ---
    {
        long n = (long)B_ * S_ * E_;
        k_embed<<<dim3((n + 255) / 256), 256, 0, stream>>>(tok, emb, x16);
    }

    // --- BiLSTM layers ---
    for (int l = 0; l < L_; ++l) {
        const _Float16* Ain  = (l == 0) ? x16 : out0_16;
        const _Float16* Wl   = Wih16 + (size_t)l * 2 * G4_ * E_;
        const float*    b0   = bih + (size_t)l * 2 * G4_;
        const float*    b1   = bhh + (size_t)l * 2 * G4_;
        // pre = Ain @ Wih[l]^T + bih[l] + bhh[l]   (M=B*S, N=2048, K=512)
        dim3 g(N2_ / 128, (B_ * S_) / 64, 1);
        k_gemm_f16<<<g, 256, 0, stream>>>(Ain, E_, 0,
                                          Wl, E_, 0,
                                          pre, N2_, 0,
                                          b0, b1,
                                          B_ * S_, N2_, E_);
        // recurrence (dir 0 fwd, dir 1 bwd)
        const _Float16* Whl = Whh16 + (size_t)l * 2 * G4_ * H_;
        k_lstm_scan<<<dim3(2), 512, 0, stream>>>(pre, Whl,
                                                 (l == L_ - 1) ? outf : nullptr,
                                                 (l == 0) ? out0_16 : out1_16);
    }

    // --- transpose final output for slot_ctx GEMM ---
    {
        long n = (long)B_ * S_ * H2_;
        k_transpose16<<<dim3((n + 255) / 256), 256, 0, stream>>>(out1_16, out_t16);
    }

    // --- attention scores: scores[b] = out[b] @ out[b]^T (batched 512x512x512) ---
    {
        dim3 g(S_ / 128, S_ / 64, B_);
        k_gemm_f16<<<g, 256, 0, stream>>>(out1_16, H2_, (long)S_ * H2_,
                                          out1_16, H2_, (long)S_ * H2_,
                                          scores, S_, (long)S_ * S_,
                                          nullptr, nullptr,
                                          S_, S_, H2_);
    }
    // --- softmax -> f16 weights ---
    k_softmax_rows<<<dim3((B_ * S_) / 8), 256, 0, stream>>>(scores, w16);

    // --- slot_ctx[b] = w[b] @ out[b]  (W-form: out^T stored (H2,S)) ---
    {
        dim3 g(H2_ / 128, S_ / 64, B_);
        k_gemm_f16<<<g, 256, 0, stream>>>(w16, S_, (long)S_ * S_,
                                          out_t16, S_, (long)H2_ * S_,
                                          slotctx, H2_, (long)S_ * H2_,
                                          nullptr, nullptr,
                                          S_, H2_, S_);
    }

    // --- intent attention, icl, gate fusion ---
    k_intent_attn<<<dim3(B_), 256, 0, stream>>>(outf, qbuf, ictx, qpi);
    k_icl<<<dim3((B_ * H2_ + 255) / 256), 256, 0, stream>>>(ictx, Wg, bg, icl);
    k_wvsum<<<dim3((H2_ + 1 + 255) / 256), 256, 0, stream>>>(Wv, bv, wvs);
    k_gate_slotin<<<dim3((B_ * S_) / 8), 256, 0, stream>>>(slotctx, icl, outf, wvs, slotin);

    // --- slot_outputs = slot_in @ Wslot^T + bslot -> d_out region 0 ---
    {
        dim3 g((SLOT_OUT_ + 127) / 128, (B_ * S_) / 64, 1);
        k_gemm_f16<<<g, 256, 0, stream>>>(slotin, H2_, 0,
                                          Wslot16, H2_, 0,
                                          slot_out, SLOT_OUT_, 0,
                                          bslot, nullptr,
                                          B_ * S_, SLOT_OUT_, H2_);
    }

    // --- intent head -> d_out region 1 ---
    k_intent_vec<<<dim3((B_ * INTENT_OUT_ + 255) / 256), 256, 0, stream>>>(qpi, Wint, bint, iv);
    {
        long n = (long)B_ * S_ * INTENT_OUT_;
        k_bcast_intent<<<dim3((n + 255) / 256), 256, 0, stream>>>(iv, intent_o);
    }

    (void)in_sizes; (void)n_in; (void)out_size; (void)ws_size;
}
